// MultiHeadAttention_9869834846848
// MI455X (gfx1250) — matrix operand order repaired and checked by reading, not compiled, not run on hardware
//
#include <hip/hip_runtime.h>
#include <hip/hip_bf16.h>

// ---------------------------------------------------------------------------
// MHA on gfx1250: bf16 WMMA (v_wmma_f32_16x16x32_bf16) for all matmuls,
// flash-style fused attention with online softmax (DPP reductions, no
// ds_bpermute), LDS staging only for the C-layout -> A-fragment P transpose.
// GEMMs use 32x32 tiles/wave for 2x register reuse of fragments.
// ---------------------------------------------------------------------------

typedef __bf16 bf16_t;
typedef __attribute__((ext_vector_type(8)))  __bf16 v8bf;
typedef __attribute__((ext_vector_type(16))) __bf16 v16bf;
typedef __attribute__((ext_vector_type(8)))  float  v8f;

#define BATCH 2
#define SEQ   2048
#define DMODEL 1024
#define NHEAD 16
#define DHEAD 64

__device__ __forceinline__ v8f wmma_bf16(v16bf a, v16bf b, v8f c) {
    // 8 args: (neg_a, A, neg_b, B, c_mod, C, reuse_a, reuse_b)
    return __builtin_amdgcn_wmma_f32_16x16x32_bf16(false, a, false, b,
                                                   (short)0, c, false, false);
}

// ---- DPP 16-lane reductions (replicated result; stays within DPP row) -----
template <int CTRL>
__device__ __forceinline__ float dpp_xfer(float x) {
    int i = __builtin_bit_cast(int, x);
    int r = __builtin_amdgcn_update_dpp(i, i, CTRL, 0xf, 0xf, true);
    return __builtin_bit_cast(float, r);
}
__device__ __forceinline__ float red_max16(float x) {
    x = fmaxf(x, dpp_xfer<0xB1>(x));   // quad_perm [1,0,3,2]  (xor 1)
    x = fmaxf(x, dpp_xfer<0x4E>(x));   // quad_perm [2,3,0,1]  (xor 2)
    x = fmaxf(x, dpp_xfer<0x141>(x));  // row_half_mirror      (other quad of 8)
    x = fmaxf(x, dpp_xfer<0x140>(x));  // row_mirror           (other 8 of 16)
    return x;
}
__device__ __forceinline__ float red_sum16(float x) {
    x += dpp_xfer<0xB1>(x);
    x += dpp_xfer<0x4E>(x);
    x += dpp_xfer<0x141>(x);
    x += dpp_xfer<0x140>(x);
    return x;
}

// A fragment: 16x32 bf16 (MxK). lane<16: row=lane, K chunks [0..7],[16..23];
// lane>=16: row=lane-16, K chunks [8..15],[24..31]. All contiguous 16B loads.
__device__ __forceinline__ v16bf load_fragA(const bf16_t* __restrict__ p,
                                            int ld, int m0, int k0) {
    int lane = threadIdx.x & 31;
    int row  = m0 + (lane & 15);
    int koff = k0 + ((lane >> 4) << 3);
    const bf16_t* base = p + (size_t)row * ld + koff;
    union { v16bf v; v8bf h[2]; } u;
    u.h[0] = *(const v8bf*)(base);
    u.h[1] = *(const v8bf*)(base + 16);
    return u.v;
}

// B fragment: 32x16 bf16 (KxN), B[k][n] = Wrow[n][k]. lane<16: N=lane,
// K chunks [0..7],[16..23]; lane>=16: N=lane-16, K chunks [8..15],[24..31]. Contiguous 16B loads.
__device__ __forceinline__ v16bf load_fragB(const bf16_t* __restrict__ p,
                                            int ld, int n0, int k0) {
    int lane = threadIdx.x & 31;
    int row  = n0 + (lane & 15);
    int koff = k0 + ((lane >> 4) << 3);
    const bf16_t* base = p + (size_t)row * ld + koff;
    union { v16bf v; v8bf h[2]; } u;
    u.h[0] = *(const v8bf*)(base);
    u.h[1] = *(const v8bf*)(base + 16);
    return u.v;
}

// ---------------------------------------------------------------------------
__global__ void cvt_f32_bf16(const float* __restrict__ in,
                             bf16_t* __restrict__ out, int n) {
    int i = blockIdx.x * blockDim.x + threadIdx.x;
    int stride = gridDim.x * blockDim.x;
    for (; i < n; i += stride) out[i] = (bf16_t)in[i];
}

// ---------------------------------------------------------------------------
// C = X @ W^T. X:[M,K] row-major bf16, W:[N,K] row-major bf16.
// One wave computes a 32x32 tile (4 WMMAs / K-chunk, 2 A + 2 B frag loads).
// MODE 0: bf16 row-major out. MODE 1: bf16 per-head transposed ([B,H,DK,S])
// for the V projection. MODE 2: f32 row-major out (final output).
template <int MODE>
__device__ __forceinline__ void store_tile16(void* __restrict__ out,
                                             int m0, int n0, v8f acc) {
    int lane = threadIdx.x & 31;
    int grp = lane >> 4, col = lane & 15;
    #pragma unroll
    for (int r = 0; r < 8; ++r) {
        int row = m0 + r + (grp << 3);
        int cn  = n0 + col;
        if (MODE == 2) {
            ((float*)out)[(size_t)row * DMODEL + cn] = acc[r];
        } else if (MODE == 0) {
            ((bf16_t*)out)[(size_t)row * DMODEL + cn] = (bf16_t)acc[r];
        } else { // MODE 1: Vt[b][h][d][s]
            int bb = row >> 11, s = row & (SEQ - 1);
            int hh = cn >> 6,  d = cn & (DHEAD - 1);
            size_t idx = (((size_t)(bb * NHEAD + hh) * DHEAD + d) << 11) + s;
            ((bf16_t*)out)[idx] = (bf16_t)acc[r];
        }
    }
}

template <int MODE>
__global__ void gemm_xwT(const bf16_t* __restrict__ X,
                         const bf16_t* __restrict__ W,
                         void* __restrict__ out) {
    const int K = DMODEL;
    int w  = threadIdx.x >> 5;
    int gw = blockIdx.x * 4 + w;            // one 32x32 tile per wave
    int nt = gw & 31;                       // N/32 = 32 tiles
    int mt = gw >> 5;                       // M/32 = 128 tiles
    int m0 = mt << 5, n0 = nt << 5;

    v8f a00 = {}, a01 = {}, a10 = {}, a11 = {};
    for (int k0 = 0; k0 < K; k0 += 32) {
        v16bf fa0 = load_fragA(X, K, m0,      k0);
        v16bf fa1 = load_fragA(X, K, m0 + 16, k0);
        v16bf fb0 = load_fragB(W, K, n0,      k0);
        v16bf fb1 = load_fragB(W, K, n0 + 16, k0);
        a00 = wmma_bf16(fa0, fb0, a00);
        a01 = wmma_bf16(fa0, fb1, a01);
        a10 = wmma_bf16(fa1, fb0, a10);
        a11 = wmma_bf16(fa1, fb1, a11);
    }
    store_tile16<MODE>(out, m0,      n0,      a00);
    store_tile16<MODE>(out, m0,      n0 + 16, a01);
    store_tile16<MODE>(out, m0 + 16, n0,      a10);
    store_tile16<MODE>(out, m0 + 16, n0 + 16, a11);
}

// ---------------------------------------------------------------------------
// Fused causal attention. One wave handles one (b, h, 16-query tile);
// key tiles of 32, online softmax. Qh/Kh: [B,S,D] bf16 (head at col h*64).
// Vt: [B,H,DK,S] bf16. O: [B,S,D] bf16.
__global__ void attn_fused(const bf16_t* __restrict__ Qh,
                           const bf16_t* __restrict__ Kh,
                           const bf16_t* __restrict__ Vt,
                           bf16_t* __restrict__ O) {
    __shared__ bf16_t pbuf[4][16 * 32];     // per-wave P staging (1 KB each)
    int lane = threadIdx.x & 31;
    int w    = threadIdx.x >> 5;
    int gw   = blockIdx.x * 4 + w;
    int qt = gw & 127;
    int h  = (gw >> 7) & (NHEAD - 1);
    int b  = gw >> 11;
    int q0 = qt << 4;
    int grp = lane >> 4, col = lane & 15;

    const bf16_t* qp = Qh + (size_t)b * SEQ * DMODEL + h * DHEAD;
    const bf16_t* kp = Kh + (size_t)b * SEQ * DMODEL + h * DHEAD;
    const bf16_t* vp = Vt + (size_t)(b * NHEAD + h) * DHEAD * SEQ;

    // Q fragments over DK=64 (two K-chunks of 32), loaded once.
    v16bf qf0 = load_fragA(qp, DMODEL, q0, 0);
    v16bf qf1 = load_fragA(qp, DMODEL, q0, 32);

    v8f o0 = {}, o1 = {}, o2 = {}, o3 = {};
    float mrow[8], lrow[8];
    #pragma unroll
    for (int r = 0; r < 8; ++r) { mrow[r] = -1e30f; lrow[r] = 0.f; }

    const float scale = 0.125f;             // 1/sqrt(64)
    int nkt = (q0 + 16 + 31) >> 5;          // key tiles needed (causal)

    for (int kt = 0; kt < nkt; ++kt) {
        int k0 = kt << 5;
        // S tile: 16 queries x 32 keys as two C tiles (keys k0.., k0+16..)
        v8f s0 = {}, s1 = {};
        s0 = wmma_bf16(qf0, load_fragB(kp, DMODEL, k0,      0),  s0);
        s0 = wmma_bf16(qf1, load_fragB(kp, DMODEL, k0,      32), s0);
        s1 = wmma_bf16(qf0, load_fragB(kp, DMODEL, k0 + 16, 0),  s1);
        s1 = wmma_bf16(qf1, load_fragB(kp, DMODEL, k0 + 16, 32), s1);

        int key0 = k0 + col, key1 = k0 + 16 + col;
        #pragma unroll
        for (int r = 0; r < 8; ++r) {
            int qrow = q0 + r + (grp << 3);
            float a0 = s0[r] * scale; if (key0 > qrow) a0 = -1e9f;
            float a1 = s1[r] * scale; if (key1 > qrow) a1 = -1e9f;
            // 16-lane reductions entirely in the VALU via DPP
            float mx = red_max16(fmaxf(a0, a1));
            float mnew = fmaxf(mrow[r], mx);
            float e0 = __expf(a0 - mnew);
            float e1 = __expf(a1 - mnew);
            float rs = red_sum16(e0 + e1);
            float corr = __expf(mrow[r] - mnew);
            lrow[r] = lrow[r] * corr + rs;
            mrow[r] = mnew;
            o0[r] *= corr; o1[r] *= corr; o2[r] *= corr; o3[r] *= corr;
            int prow = r + (grp << 3);
            pbuf[w][prow * 32 + col]      = (bf16_t)e0;
            pbuf[w][prow * 32 + 16 + col] = (bf16_t)e1;
        }

        // Re-load P as an A-fragment (16x32) from LDS (per-wave private).
        {
            const bf16_t* pb = &pbuf[w][(lane & 15) * 32 + ((lane >> 4) << 3)];
            union { v16bf v; v8bf hh[2]; } u;
            u.hh[0] = *(const v8bf*)(pb);
            u.hh[1] = *(const v8bf*)(pb + 16);
            v16bf pf = u.v;
            // O += P @ V  (Vt rows = d, contiguous keys)
            o0 = wmma_bf16(pf, load_fragB(vp, SEQ, 0,  k0), o0);
            o1 = wmma_bf16(pf, load_fragB(vp, SEQ, 16, k0), o1);
            o2 = wmma_bf16(pf, load_fragB(vp, SEQ, 32, k0), o2);
            o3 = wmma_bf16(pf, load_fragB(vp, SEQ, 48, k0), o3);
        }
    }

    // Epilogue: divide by l, store bf16 heads back into [B,S,D] layout.
    #pragma unroll
    for (int r = 0; r < 8; ++r) {
        int row = q0 + r + (grp << 3);
        float inv = 1.0f / lrow[r];
        size_t base = ((size_t)b * SEQ + row) * DMODEL + h * DHEAD;
        O[base + 0  + col] = (bf16_t)(o0[r] * inv);
        O[base + 16 + col] = (bf16_t)(o1[r] * inv);
        O[base + 32 + col] = (bf16_t)(o2[r] * inv);
        O[base + 48 + col] = (bf16_t)(o3[r] * inv);
    }
}

// ---------------------------------------------------------------------------
extern "C" void kernel_launch(void* const* d_in, const int* in_sizes, int n_in,
                              void* d_out, int out_size, void* d_ws, size_t ws_size,
                              hipStream_t stream) {
    const float* q  = (const float*)d_in[0];
    const float* k  = (const float*)d_in[1];
    const float* v  = (const float*)d_in[2];
    // d_in[3] = mask (causal tril) -- computed analytically in-kernel
    const float* wq = (const float*)d_in[4];
    const float* wk = (const float*)d_in[5];
    const float* wv = (const float*)d_in[6];
    const float* wo = (const float*)d_in[7];

    const int NQKV = BATCH * SEQ * DMODEL;   // 4,194,304
    const int NW   = DMODEL * DMODEL;        // 1,048,576

    bf16_t* ws = (bf16_t*)d_ws;
    bf16_t* qb  = ws;                 // [B,S,D]
    bf16_t* kb  = qb  + NQKV;
    bf16_t* vb  = kb  + NQKV;
    bf16_t* wqb = vb  + NQKV;         // [D,D] x4
    bf16_t* wkb = wqb + NW;
    bf16_t* wvb = wkb + NW;
    bf16_t* wob = wvb + NW;
    bf16_t* Qh  = wob + NW;           // [B,S,D]
    bf16_t* Kh  = Qh  + NQKV;
    bf16_t* Vt  = Kh  + NQKV;         // [B,H,DK,S]
    bf16_t* At  = Vt  + NQKV;         // [B,S,D] attention output

    // Stage 1: f32 -> bf16 casts
    cvt_f32_bf16<<<2048, 256, 0, stream>>>(q,  qb,  NQKV);
    cvt_f32_bf16<<<2048, 256, 0, stream>>>(k,  kb,  NQKV);
    cvt_f32_bf16<<<2048, 256, 0, stream>>>(v,  vb,  NQKV);
    cvt_f32_bf16<<<1024, 256, 0, stream>>>(wq, wqb, NW);
    cvt_f32_bf16<<<1024, 256, 0, stream>>>(wk, wkb, NW);
    cvt_f32_bf16<<<1024, 256, 0, stream>>>(wv, wvb, NW);
    cvt_f32_bf16<<<1024, 256, 0, stream>>>(wo, wob, NW);

    // Stage 2/3: projections (128 x 32 tiles of 32x32, 4 waves/block)
    gemm_xwT<0><<<1024, 128, 0, stream>>>(qb, wqb, Qh);
    gemm_xwT<0><<<1024, 128, 0, stream>>>(kb, wkb, Kh);
    gemm_xwT<1><<<1024, 128, 0, stream>>>(vb, wvb, Vt);

    // Stage 4: fused causal attention (4096 waves -> 1024 blocks)
    attn_fused<<<1024, 128, 0, stream>>>(Qh, Kh, Vt, At);

    // Stage 5: output projection, f32 store to d_out
    gemm_xwT<2><<<1024, 128, 0, stream>>>(At, wob, (void*)d_out);
}
